// GATEncoder_50955492000114
// MI455X (gfx1250) — compile-verified, matrix operand order checked by reading, not run on hardware
//
#include <hip/hip_runtime.h>
#include <hip/hip_bf16.h>
#include <math.h>

typedef __attribute__((ext_vector_type(16))) _Float16 v16h;
typedef __attribute__((ext_vector_type(8)))  float    v8f;

#define HEADS     2
#define HID       64
#define CH        128   // HEADS*HID
#define IN_DIM    128
#define NEG_SLOPE 0.2f

// ---- order-preserving float<->uint encoding for atomic max ----------------
__device__ __forceinline__ unsigned enc_f32(float f) {
  unsigned u = __float_as_uint(f);
  return (u & 0x80000000u) ? ~u : (u | 0x80000000u);
}
__device__ __forceinline__ float dec_f32(unsigned u) {
  unsigned b = (u & 0x80000000u) ? (u & 0x7FFFFFFFu) : ~u;
  return __uint_as_float(b);
}

// ---- init: zero the aggregation target + softmax denominators, m = -inf ---
__global__ void gat_init(float* __restrict__ out, float* __restrict__ ssum,
                         unsigned* __restrict__ mx, int n) {
  int i = blockIdx.x * blockDim.x + threadIdx.x;
  if (i < n * CH) out[i] = 0.0f;
  if (i < n * HEADS) { ssum[i] = 0.0f; mx[i] = 0x007FFFFFu; /* enc(-inf) */ }
}

// ---- GEMM: h = x @ W  via v_wmma_f32_16x16x32_f16 -------------------------
// block = 256 threads = 8 waves; block handles a 32-row M stripe, each wave
// owns one 16-col N tile and computes TWO stacked 16x16 C tiles (B reused).
// K loop fully unrolled: 8 wmma per wave.
__global__ void __launch_bounds__(256) gat_gemm_wmma(
    const float* __restrict__ x, const float* __restrict__ W,
    float* __restrict__ h, int n) {
  __shared__ _Float16 xs[32 * 136];          // padded row stride vs bank clash
  const int mt   = blockIdx.x;               // 32-row stripe index
  const int tid  = threadIdx.x;
  const int wave = tid >> 5;
  const int lane = tid & 31;

  // cooperative f32->f16 stage of the 32x128 A stripe
  #pragma unroll
  for (int i = 0; i < 16; ++i) {
    int idx = i * 256 + tid;
    int row = idx >> 7, col = idx & 127;
    int gr  = mt * 32 + row;
    float v = (gr < n) ? x[(size_t)gr * IN_DIM + col] : 0.0f;
    xs[row * 136 + col] = (_Float16)v;
  }
  __syncthreads();

  const int nt   = wave;                    // N tile 0..7
  const int col  = nt * 16 + (lane & 15);
  const int mrow = lane & 15;
  const int hi   = (lane & 16) >> 1;        // +8 K offset for upper half-wave
  v8f c0 = {}, c1 = {};

  #pragma unroll
  for (int kb = 0; kb < 4; ++kb) {
    v16h a0, a1, b;
    #pragma unroll
    for (int e = 0; e < 16; ++e) {
      // ISA 16-bit A 16x32 layout: K = (e&7) + (lane>=16 ? 8:0) + (e>=8 ? 16:0)
      int kl = (e & 7) + hi + ((e & 8) << 1);
      int k  = kb * 32 + kl;
      a0[e] = xs[mrow * 136 + k];
      a1[e] = xs[(16 + mrow) * 136 + k];
      b[e]  = (_Float16)W[(size_t)k * CH + col];  // B: lane = column, same K map
    }
    c0 = __builtin_amdgcn_wmma_f32_16x16x32_f16(false, a0, false, b,
                                                (short)0, c0, false, false);
    c1 = __builtin_amdgcn_wmma_f32_16x16x32_f16(false, a1, false, b,
                                                (short)0, c1, false, false);
  }

  // C/D layout: VGPR r -> row r (lanes 0-15) / row r+8 (lanes 16-31)
  const int base = mt * 32;
  if (base + 31 < n) {                      // full-stripe fast path (no guards)
    #pragma unroll
    for (int r = 0; r < 8; ++r) {
      h[(size_t)(base + r + hi) * CH + col]      = c0[r];
      h[(size_t)(base + 16 + r + hi) * CH + col] = c1[r];
    }
  } else {
    #pragma unroll
    for (int r = 0; r < 8; ++r) {
      int r0 = base + r + hi, r1 = base + 16 + r + hi;
      if (r0 < n) h[(size_t)r0 * CH + col] = c0[r];
      if (r1 < n) h[(size_t)r1 * CH + col] = c1[r];
    }
  }
}

// ---- per-node attention scores: a_s[n,h], a_d[n,h] ------------------------
__global__ void __launch_bounds__(256) gat_scores(
    const float* __restrict__ h, const float* __restrict__ att_src,
    const float* __restrict__ att_dst, float* __restrict__ a_s,
    float* __restrict__ a_d, int n) {
  int gw   = (blockIdx.x * blockDim.x + threadIdx.x) >> 5;
  int lane = threadIdx.x & 31;
  if (gw >= n) return;
  const float* hp = h + (size_t)gw * CH;
  float v0 = hp[lane], v1 = hp[lane + 32], v2 = hp[lane + 64], v3 = hp[lane + 96];
  float s0 = v0 * att_src[lane]      + v1 * att_src[lane + 32];
  float s1 = v2 * att_src[lane + 64] + v3 * att_src[lane + 96];
  float d0 = v0 * att_dst[lane]      + v1 * att_dst[lane + 32];
  float d1 = v2 * att_dst[lane + 64] + v3 * att_dst[lane + 96];
  #pragma unroll
  for (int o = 16; o; o >>= 1) {
    s0 += __shfl_xor(s0, o, 32);
    s1 += __shfl_xor(s1, o, 32);
    d0 += __shfl_xor(d0, o, 32);
    d1 += __shfl_xor(d1, o, 32);
  }
  if (lane == 0) {
    a_s[gw * 2 + 0] = s0; a_s[gw * 2 + 1] = s1;
    a_d[gw * 2 + 0] = d0; a_d[gw * 2 + 1] = d1;
  }
}

__device__ __forceinline__ void edge_endpoints(const int* __restrict__ ei,
                                               int e, int E, int& s, int& d) {
  if (e < E) { s = ei[e]; d = ei[E + e]; }
  else       { s = d = e - E; }             // self loops appended
}

// ---- edge pass 1: segment max of leaky-relu logits ------------------------
__global__ void gat_edge_max(const int* __restrict__ ei,
                             const float* __restrict__ a_s,
                             const float* __restrict__ a_d,
                             unsigned* __restrict__ mx, int E, int n) {
  int e = blockIdx.x * blockDim.x + threadIdx.x;
  if (e >= E + n) return;
  int s, d; edge_endpoints(ei, e, E, s, d);
  #pragma unroll
  for (int hh = 0; hh < HEADS; ++hh) {
    float lg = a_s[s * 2 + hh] + a_d[d * 2 + hh];
    lg = lg >= 0.0f ? lg : NEG_SLOPE * lg;
    atomicMax(&mx[d * 2 + hh], enc_f32(lg));
  }
}

// ---- edge pass 2: segment sum of exp(logit - max) -------------------------
__global__ void gat_edge_sum(const int* __restrict__ ei,
                             const float* __restrict__ a_s,
                             const float* __restrict__ a_d,
                             const unsigned* __restrict__ mx,
                             float* __restrict__ ssum, int E, int n) {
  int e = blockIdx.x * blockDim.x + threadIdx.x;
  if (e >= E + n) return;
  int s, d; edge_endpoints(ei, e, E, s, d);
  #pragma unroll
  for (int hh = 0; hh < HEADS; ++hh) {
    float lg = a_s[s * 2 + hh] + a_d[d * 2 + hh];
    lg = lg >= 0.0f ? lg : NEG_SLOPE * lg;
    float ex = expf(lg - dec_f32(mx[d * 2 + hh]));
    atomicAdd(&ssum[d * 2 + hh], ex);
  }
}

// ---- edge pass 3: out[dst] += alpha * h[src]; one wave per edge -----------
__global__ void __launch_bounds__(256) gat_edge_aggr(
    const int* __restrict__ ei, const float* __restrict__ a_s,
    const float* __restrict__ a_d, const unsigned* __restrict__ mx,
    const float* __restrict__ ssum, const float* __restrict__ h,
    float* __restrict__ out, int E, int n) {
  int gw   = (blockIdx.x * blockDim.x + threadIdx.x) >> 5;
  int lane = threadIdx.x & 31;
  if (gw >= E + n) return;
  int s, d; edge_endpoints(ei, gw, E, s, d);
  float al[HEADS];
  #pragma unroll
  for (int hh = 0; hh < HEADS; ++hh) {
    float lg = a_s[s * 2 + hh] + a_d[d * 2 + hh];
    lg = lg >= 0.0f ? lg : NEG_SLOPE * lg;
    float ex = expf(lg - dec_f32(mx[d * 2 + hh]));
    al[hh] = ex / (ssum[d * 2 + hh] + 1e-16f);
  }
  const float* hs = h   + (size_t)s * CH;
  float*       op = out + (size_t)d * CH;
  #pragma unroll
  for (int j = 0; j < 4; ++j) {            // channels lane+32j; head = j>>1
    int c = lane + 32 * j;
    atomicAdd(&op[c], hs[c] * al[j >> 1]);
  }
}

// ---- epilogue: out = PReLU(out + bias) ------------------------------------
__global__ void gat_finalize(float* __restrict__ out,
                             const float* __restrict__ bias,
                             const float* __restrict__ prelu_w, int tot) {
  int i = blockIdx.x * blockDim.x + threadIdx.x;
  if (i >= tot) return;
  int c = i & (CH - 1);
  float v = out[i] + bias[c];
  out[i] = v >= 0.0f ? v : prelu_w[c] * v;
}

extern "C" void kernel_launch(void* const* d_in, const int* in_sizes, int n_in,
                              void* d_out, int out_size, void* d_ws, size_t ws_size,
                              hipStream_t stream) {
  const float* x       = (const float*)d_in[0];
  const int*   ei      = (const int*)d_in[1];
  const float* W       = (const float*)d_in[2];
  const float* att_src = (const float*)d_in[3];
  const float* att_dst = (const float*)d_in[4];
  const float* bias    = (const float*)d_in[5];
  const float* prelu_w = (const float*)d_in[6];
  float* out = (float*)d_out;

  const int n  = in_sizes[0] / IN_DIM;
  const int E  = in_sizes[1] / 2;
  const int Et = E + n;
  const int tot = n * CH;

  // workspace layout (all fits in L2 on MI455X): h | a_s | a_d | ssum | mx
  float*    h    = (float*)d_ws;
  float*    a_s  = h   + (size_t)n * CH;
  float*    a_d  = a_s + (size_t)n * HEADS;
  float*    ssum = a_d + (size_t)n * HEADS;
  unsigned* mx   = (unsigned*)(ssum + (size_t)n * HEADS);

  gat_init<<<(tot + 255) / 256, 256, 0, stream>>>(out, ssum, mx, n);
  gat_gemm_wmma<<<(n + 31) / 32, 256, 0, stream>>>(x, W, h, n);
  gat_scores<<<(n + 7) / 8, 256, 0, stream>>>(h, att_src, att_dst, a_s, a_d, n);
  gat_edge_max<<<(Et + 255) / 256, 256, 0, stream>>>(ei, a_s, a_d, mx, E, n);
  gat_edge_sum<<<(Et + 255) / 256, 256, 0, stream>>>(ei, a_s, a_d, mx, ssum, E, n);
  gat_edge_aggr<<<(Et + 7) / 8, 256, 0, stream>>>(ei, a_s, a_d, mx, ssum, h, out, E, n);
  gat_finalize<<<(tot + 255) / 256, 256, 0, stream>>>(out, bias, prelu_w, tot);
}
